// WindowAttention_5471788335510
// MI455X (gfx1250) — matrix operand order repaired and checked by reading, not compiled, not run on hardware
//
#include <hip/hip_runtime.h>
#include <hip/hip_bf16.h>

// Window attention (Swin-style), MI455X / gfx1250, wave32 + WMMA f16->f32.
// Pipeline:
//   k0: convert w_qkv -> wqkvT (f16, [288][96]), w_out -> woutT (f16, [96][96])
//   k1: per (batch, window): x gather->f16 LDS, QKV GEMM (WMMA), S=QK^T (WMMA),
//       softmax+bias (VALU), O=PV (WMMA) -> y f16 [b][3136][96] in scratch
//   k2: out = y @ w_out^T + b_out, WMMA, f32 stores transposed to [b][96][56][56]

typedef _Float16 half8 __attribute__((ext_vector_type(8)));
typedef _Float16 v16h  __attribute__((ext_vector_type(16)));
typedef float    v8f   __attribute__((ext_vector_type(8)));

__device__ __forceinline__ v8f wmma_f16(v16h a, v16h b, v8f c) {
  // (neg_a, A, neg_b, B, c_mod, C, reuse_a, reuse_b)
  return __builtin_amdgcn_wmma_f32_16x16x32_f16(false, a, false, b, (short)0, c,
                                                false, false);
}

// Load 16 halfs as two 16B chunks: chunk0 at p, chunk1 at p+off2 halfs.
// A operand: off2 = 8 (contiguous 32B, K = 16*half + 0..15);
// B operand (transposed [N][K] storage): off2 = 8 (contiguous 32B).
__device__ __forceinline__ v16h ld_pair(const _Float16* p, int off2) {
  v16h r;
  *(half8*)&r       = *(const half8*)p;
  *((half8*)&r + 1) = *(const half8*)(p + off2);
  return r;
}

// ---------------------------------------------------------------- kernel 0
__global__ __launch_bounds__(256) void cvt_weights(
    const float* __restrict__ w_qkv,   // [96][288]
    const float* __restrict__ w_out,   // [96][96]
    _Float16* __restrict__ wqkvT,      // [288][96]
    _Float16* __restrict__ woutT) {    // [96][96] (transposed)
  int i = blockIdx.x * 256 + threadIdx.x;
  if (i < 288 * 96) {
    int n = i / 96, k = i % 96;
    wqkvT[i] = (_Float16)w_qkv[k * 288 + n];
  } else if (i < 288 * 96 + 96 * 96) {
    int j = i - 288 * 96;
    int n = j / 96, k = j % 96;
    woutT[j] = (_Float16)w_out[k * 96 + n];
  }
}

// ---------------------------------------------------------------- kernel 1
// LDS map (bytes), all b128 bases 16B aligned, row strides chosen so the
// 16-lane row sweep of A/B tile loads steps a non-multiple-of-64 bank count:
//   [0      , 37888) qkv   f16 [64][296]   (row 592B)
//   [37888  , 55296) S/P   f32 [64][68] / f16 [64][136] (row 272B)
//                     (x-tile Ax f16 [64][104] aliases here; dead before S)
//   [55296  , 59904) Vt    f16 [32][72]    (row 144B)
//   [59904  , 61952) pos_emb f32 [507]
__global__ __launch_bounds__(256) void win_attn(
    const float* __restrict__ x,        // [64][96][56][56]
    const float* __restrict__ pos_emb,  // [169][3]
    const _Float16* __restrict__ wqkvT, // [288][96]
    _Float16* __restrict__ y) {         // [64][3136][96]
  __shared__ __align__(16) char lds[61952];
  _Float16* qkv  = (_Float16*)lds;
  float*    Sf   = (float*)(lds + 37888);
  _Float16* Pf   = (_Float16*)(lds + 37888);
  _Float16* Ax   = (_Float16*)(lds + 37888);
  _Float16* Vt   = (_Float16*)(lds + 55296);
  float*    posL = (float*)(lds + 59904);

  const int tid  = threadIdx.x;
  const int lane = tid & 31;
  const int wav  = tid >> 5;    // 8 waves
  const int hi   = lane >> 4;   // lane half selects K sub-chunk
  const int ln   = lane & 15;

  const int blk = blockIdx.x;   // 4096 = 64 batches * 64 windows
  const int b   = blk >> 6;
  const int w   = blk & 63;
  const int wy  = w >> 3, wx = w & 7;

  for (int i = tid; i < 507; i += 256) posL[i] = pos_emb[i];

  // Gather x window -> Ax f16 [64 tokens][96 ch], zero-pad tokens 49..63.
  const float* xb = x + (size_t)b * 96 * 3136;
  for (int i = tid; i < 64 * 96; i += 256) {
    int m = i & 63, k = i >> 6;          // consecutive threads: consecutive m
    _Float16 v = (_Float16)0.f;
    if (m < 49) {
      int py = wy * 7 + m / 7, px = wx * 7 + m % 7;
      v = (_Float16)xb[(size_t)k * 3136 + py * 56 + px];
    }
    Ax[m * 104 + k] = v;
  }
  __syncthreads();

  // QKV GEMM: [64x96] @ [96x288] -> qkv f16.  72 tiles / 8 waves.
  for (int t = wav; t < 72; t += 8) {
    int mT = t & 3, nT = t >> 2;
    v8f acc = {};
#pragma unroll
    for (int kT = 0; kT < 3; ++kT) {
      v16h a  = ld_pair(Ax + (mT * 16 + ln) * 104 + kT * 32 + hi * 16, 8);
      v16h bm = ld_pair(wqkvT + (size_t)(nT * 16 + ln) * 96 + kT * 32 + hi * 16, 8);
      acc = wmma_f16(a, bm, acc);
    }
#pragma unroll
    for (int i = 0; i < 8; ++i) {
      int m = mT * 16 + i + hi * 8;
      qkv[m * 296 + nT * 16 + ln] = (_Float16)acc[i];
    }
  }
  __syncthreads();

  const float scale = 0.17677669529663687f;  // 32^-0.5

  for (int h = 0; h < 3; ++h) {
    // V^T for this head: Vt[d][j] = qkv[j][192 + h*32 + d]
    for (int i = tid; i < 32 * 64; i += 256) {
      int d = i >> 6, j = i & 63;
      Vt[d * 72 + j] = qkv[j * 296 + 192 + h * 32 + d];
    }
    // S = Q K^T * scale.  A = Q rows of qkv; B^T = K rows of qkv (natural).
    for (int t = wav; t < 16; t += 8) {
      int mT = t & 3, nT = t >> 2;
      v16h a  = ld_pair(qkv + (mT * 16 + ln) * 296 + h * 32 + hi * 16, 8);
      v16h bm = ld_pair(qkv + (nT * 16 + ln) * 296 + 96 + h * 32 + hi * 16, 8);
      v8f acc = {};
      acc = wmma_f16(a, bm, acc);
#pragma unroll
      for (int i = 0; i < 8; ++i)
        Sf[(mT * 16 + i + hi * 8) * 68 + nT * 16 + ln] = acc[i] * scale;
    }
    __syncthreads();

    // Softmax + relative position bias; write P (f16) in-place over S rows.
    // rel(i,j) = (xj - xi + 6)*13 + (yj - yi + 6), p = x*7 + y.
    if (tid < 64) {
      int i = tid;
      _Float16* Prow = Pf + i * 136;
      if (i < 49) {
        float* Srow = Sf + i * 68;
        int xi = i / 7, yi = i % 7;
        float mx = -1e30f;
        for (int j = 0; j < 49; ++j) {
          int rel = (j / 7 - xi + 6) * 13 + (j % 7 - yi + 6);
          mx = fmaxf(mx, Srow[j] + posL[rel * 3 + h]);
        }
        float sum = 0.f;
        for (int j = 0; j < 49; ++j) {
          int rel = (j / 7 - xi + 6) * 13 + (j % 7 - yi + 6);
          float e = __expf(Srow[j] + posL[rel * 3 + h] - mx);
          sum += e;
          Prow[j] = (_Float16)e;  // clobbers Srow[j/2], already consumed
        }
        float r = 1.f / sum;
        for (int j = 0; j < 49; ++j) Prow[j] = (_Float16)((float)Prow[j] * r);
        for (int j = 49; j < 64; ++j) Prow[j] = (_Float16)0.f;
      } else {
        for (int j = 0; j < 64; ++j) Prow[j] = (_Float16)0.f;  // pad rows
      }
    }
    __syncthreads();

    // O = P @ V : 8 tiles (4 mT x 2 nT), one per wave. K = 64 (2 steps).
    {
      int mT = wav & 3, nT = wav >> 2;
      v8f acc = {};
#pragma unroll
      for (int kT = 0; kT < 2; ++kT) {
        v16h a  = ld_pair(Pf + (mT * 16 + ln) * 136 + kT * 32 + hi * 16, 8);
        v16h bm = ld_pair(Vt + (nT * 16 + ln) * 72 + kT * 32 + hi * 16, 8);
        acc = wmma_f16(a, bm, acc);
      }
      int col = h * 32 + nT * 16 + ln;
#pragma unroll
      for (int i = 0; i < 8; ++i) {
        int m = mT * 16 + i + hi * 8;
        if (m < 49) {
          int py = wy * 7 + m / 7, px = wx * 7 + m % 7;
          y[((size_t)b * 3136 + py * 56 + px) * 96 + col] = (_Float16)acc[i];
        }
      }
    }
    __syncthreads();  // SP / Vt reuse next head
  }
}

// ---------------------------------------------------------------- kernel 2
// out = y @ w_out^T + b_out, stored as [b][96][3136] f32.
// 3136 blocks: 64 rows of y x all 96 out channels each (24 tiles / 8 waves).
__global__ __launch_bounds__(256) void out_proj(
    const _Float16* __restrict__ y,     // [64][3136][96]
    const _Float16* __restrict__ woutT, // [96][96] transposed
    const float* __restrict__ b_out,    // [96]
    float* __restrict__ out) {          // [64][96][3136]
  const int tid = threadIdx.x, lane = tid & 31, wav = tid >> 5;
  const int hi = lane >> 4, ln = lane & 15;
  const int blk = blockIdx.x;          // 64 * 49
  const int b = blk / 49, rb = blk % 49;
  const int pixBase = rb * 64;
  const _Float16* yb = y + ((size_t)b * 3136 + pixBase) * 96;
  float* ob = out + (size_t)b * 96 * 3136;

  for (int t = wav; t < 24; t += 8) {
    int mT = t & 3, nT = t >> 2;       // nT 0..5
    v8f acc = {};
#pragma unroll
    for (int kT = 0; kT < 3; ++kT) {
      v16h a  = ld_pair(yb + (mT * 16 + ln) * 96 + kT * 32 + hi * 16, 8);
      v16h bm = ld_pair(woutT + (nT * 16 + ln) * 96 + kT * 32 + hi * 16, 8);
      acc = wmma_f16(a, bm, acc);
    }
    int n = nT * 16 + ln;
    float bias = b_out[n];
    // lane holds 8 consecutive pixels (m = mT*16 + hi*8 + 0..7) at fixed n.
    float* dst = ob + (size_t)n * 3136 + pixBase + mT * 16 + hi * 8;
    float4 lo = make_float4(acc[0] + bias, acc[1] + bias, acc[2] + bias, acc[3] + bias);
    float4 hh = make_float4(acc[4] + bias, acc[5] + bias, acc[6] + bias, acc[7] + bias);
    *(float4*)dst       = lo;
    *(float4*)(dst + 4) = hh;
  }
}

// ---------------------------------------------------------------- launch
extern "C" void kernel_launch(void* const* d_in, const int* in_sizes, int n_in,
                              void* d_out, int out_size, void* d_ws, size_t ws_size,
                              hipStream_t stream) {
  const float* x       = (const float*)d_in[0];  // [64,96,56,56]
  const float* w_qkv   = (const float*)d_in[1];  // [96,288]
  const float* pos_emb = (const float*)d_in[2];  // [169,3]
  const float* w_out   = (const float*)d_in[3];  // [96,96]
  const float* b_out   = (const float*)d_in[4];  // [96]
  float* out = (float*)d_out;                    // [64,96,56,56]

  char* ws = (char*)d_ws;
  // scratch map: wqkvT f16 55296B | woutT f16 18432B | y f16 38,535,168B
  _Float16* wqkvT = (_Float16*)(ws);
  _Float16* woutT = (_Float16*)(ws + 55296);
  _Float16* y     = (_Float16*)(ws + 73728);

  cvt_weights<<<144, 256, 0, stream>>>(w_qkv, w_out, wqkvT, woutT);
  win_attn<<<64 * 64, 256, 0, stream>>>(x, pos_emb, wqkvT, y);
  out_proj<<<64 * 49, 256, 0, stream>>>(y, woutT, b_out, out);
}
